// MojoQwen3MoeBlock_76673756168526
// MI455X (gfx1250) — matrix operand order repaired and checked by reading, not compiled, not run on hardware
//
#include <hip/hip_runtime.h>
#include <cstdint>
#include <cstddef>

// ---------------- CDNA5 (gfx1250) MoE transformer block ----------------
// wave32, WMMA bf16 (v_wmma_f32_16x16x32_bf16), LDS-staged tiled GEMMs with
// double-buffered Tensor-Data-Mover (tensor_load_to_lds) weight streaming,
// flash attention (1 wave / 16-query tile / head), grouped expert GEMM.

typedef __attribute__((ext_vector_type(16))) __bf16 v16bf;
typedef __attribute__((ext_vector_type(8)))  __bf16 v8bf;
typedef __attribute__((ext_vector_type(4)))  __bf16 v4bf;
typedef __attribute__((ext_vector_type(8)))  float  v8f;
typedef __attribute__((ext_vector_type(4)))  float  v4f;
typedef __attribute__((ext_vector_type(4)))  unsigned int v4u;
typedef __attribute__((ext_vector_type(8)))  int    v8i;
typedef __attribute__((ext_vector_type(4)))  int    v4i;

#define SEQ    1024
#define HID    4096
#define NHEADS 32
#define HDIM   128
#define NEXP   8
#define TOPK   2
#define QKVW   (3 * HID)      // 12288
#define NSLOT  (SEQ * TOPK)   // 2048
#define EPSF   1e-6f
#define NEGINF -1e9f

// ---- GEMM tiling ----
#define BM 64
#define BN 128
#define BK 32
#define APAD 40    // bf16 elems per LDS A row (16B-aligned, bank-conflict-free)
#define BPAD 40
#define KPAD 136   // attention K-tile row pad
#define VPAD 40
#define PPAD 40

// Tensor Data Mover availability (probe-verified builtin on ROCm 7.2 clang-22:
// 5-arg form; amdgpu-toolchain clang-23: 6-arg form).
#if defined(__has_builtin)
#if __has_builtin(__builtin_amdgcn_tensor_load_to_lds) && \
    __has_builtin(__builtin_amdgcn_s_wait_tensorcnt)
#define HAVE_TDM 1
#endif
#endif
#ifndef HAVE_TDM
#define HAVE_TDM 0
#endif

// ===================== helpers =====================
__device__ __forceinline__ unsigned pack2bf(float a, float b) {
    union { unsigned u; __bf16 h[2]; } p;
    p.h[0] = (__bf16)a; p.h[1] = (__bf16)b;
    return p.u;
}

// A-matrix 16x32 bf16 frag: lane L(0-15): row=L, K=0..7 & 16..23;
//                           lane L+16  : row=L, K=8..15 & 24..31.
__device__ __forceinline__ v16bf load_a_frag(const __bf16* lds, int row, int ldp) {
    const int lane = threadIdx.x & 31;
    const __bf16* p = lds + (size_t)(row + (lane & 15)) * ldp + ((lane >> 4) * 8);
    v8bf lo = *(const v8bf*)(p);
    v8bf hi = *(const v8bf*)(p + 16);
    return __builtin_shufflevector(lo, hi, 0,1,2,3,4,5,6,7,8,9,10,11,12,13,14,15);
}

// B-matrix 32x16 bf16 frag (K-major per lane): lane L(0-15): col=L, K=0..7 & 16..23;
//                                              lane L+16  : col=L, K=8..15 & 24..31.
// LDS holds B transposed: lds[n][k].
__device__ __forceinline__ v16bf load_b_frag(const __bf16* lds, int col, int ldp) {
    const int lane = threadIdx.x & 31;
    const __bf16* p = lds + (size_t)(col + (lane & 15)) * ldp + ((lane >> 4) * 8);
    v8bf lo = *(const v8bf*)(p);
    v8bf hi = *(const v8bf*)(p + 16);
    return __builtin_shufflevector(lo, hi, 0,1,2,3,4,5,6,7,8,9,10,11,12,13,14,15);
}

__device__ __forceinline__ v8f wmma_bf16(v16bf a, v16bf b, v8f c) {
    return __builtin_amdgcn_wmma_f32_16x16x32_bf16(false, a, false, b, (short)0, c,
                                                   false, false);
}

__device__ __forceinline__ float rmax16(float x) {
#pragma unroll
    for (int m = 1; m <= 8; m <<= 1) x = fmaxf(x, __shfl_xor(x, m, 32));
    return x;
}
__device__ __forceinline__ float rsum16(float x) {
#pragma unroll
    for (int m = 1; m <= 8; m <<= 1) x += __shfl_xor(x, m, 32);
    return x;
}

#if HAVE_TDM
// Issue a TDM load of a BKxBN f32 tile (rows x cols) of a row-major tensor with
// row length n_len (elements) into LDS. Wave-uniform; issue from one wave only.
__device__ __forceinline__ void tdm_load_b_tile(const float* gsrc, float* lds_dst,
                                                int n_len, int k_len) {
    const unsigned lds_lo = (unsigned)(uintptr_t)lds_dst;   // flat[31:0] == LDS offset
    const uint64_t ga = (uint64_t)(uintptr_t)gsrc;
    v4u g0;
    g0[0] = 1u;                                    // count=1, user descriptor
    g0[1] = lds_lo;                                // lds_addr (bytes)
    g0[2] = (unsigned)(ga & 0xffffffffu);          // global_addr[31:0]
    g0[3] = (unsigned)((ga >> 32) & 0x01ffffffu)   // global_addr[56:32]
            | 0x80000000u;                         // type=2 ("image")
    v8i g1;
    g1[0] = 2 << 16;                               // data_size=4B
    g1[1] = (int)(((unsigned)n_len & 0xffffu) << 16);        // tensor_dim0[15:0]
    g1[2] = (int)(((unsigned)n_len >> 16) & 0xffffu)         // tensor_dim0[31:16]
            | (int)(((unsigned)k_len & 0xffffu) << 16);      // tensor_dim1[15:0]
    g1[3] = (int)(((unsigned)k_len >> 16) & 0xffffu)         // tensor_dim1[31:16]
            | (BN << 16);                          // tile_dim0 = 128
    g1[4] = BK;                                    // tile_dim1 = 32, tile_dim2 = 0
    g1[5] = n_len;                                 // tensor_dim0_stride[31:0]
    g1[6] = 0;
    g1[7] = 0;
    v4i z4 = {0, 0, 0, 0};
#if defined(__clang_major__) && __clang_major__ >= 23
    v8i z8 = {0, 0, 0, 0, 0, 0, 0, 0};
    __builtin_amdgcn_tensor_load_to_lds(g0, g1, z4, z4, z8, 0);
#else
    __builtin_amdgcn_tensor_load_to_lds(g0, g1, z4, z4, 0);
#endif
}
#endif

// ===================== embedding gather =====================
__global__ void embed_kernel(const int* __restrict__ ids,
                             const float* __restrict__ table,
                             float* __restrict__ x) {
    const int row = blockIdx.x;
    const v4f* src = (const v4f*)(table + (size_t)ids[row] * HID);
    v4f* dst = (v4f*)(x + (size_t)row * HID);
    for (int i = threadIdx.x; i < HID / 4; i += blockDim.x) dst[i] = src[i];
}

// ===================== dense tiled GEMM: C[M,N] = A[M,K] @ B[K,N] ==================
// K, N compile-time -> immediate-offset addressing. B panels streamed by the
// Tensor Data Mover with double buffering (DMA of tile i+1 overlaps WMMA of tile i).
template <int K, int N>
__global__ __launch_bounds__(256) void gemm_bf16_kernel(const float* __restrict__ A,
                                                        const float* __restrict__ B,
                                                        float* __restrict__ C) {
    __shared__ __bf16 As[BM * APAD];
    __shared__ __bf16 Bs[BN * BPAD];
#if HAVE_TDM
    __shared__ float Bf32[2 * BK * BN];  // double-buffered TDM landing tiles (32 KB)
#endif
    const int tid  = threadIdx.x;
    const int lane = tid & 31;
    const int wave = tid >> 5;
    const int wm = wave >> 2;            // 0..1  (32 rows each)
    const int wn = wave & 3;             // 0..3  (32 cols each)
    const int mbase = blockIdx.y * BM;
    const int nbase = blockIdx.x * BN;
    const int half = lane >> 4, nlan = lane & 15;
    constexpr int NT = K / BK;

    v8f acc[2][2] = {};

#if HAVE_TDM
    if (wave == 0) tdm_load_b_tile(B + nbase, &Bf32[0], N, K);   // prologue: tile 0
#endif
    for (int it = 0; it < NT; ++it) {
        const int k0 = it * BK;
        // ---- stage A (64x32): 8 f32 per thread, one b128 LDS store ----
        {
            const int r  = tid >> 2;
            const int kc = (tid & 3) * 8;
            const float* ga = A + (size_t)(mbase + r) * K + k0 + kc;
            __builtin_prefetch(ga + BK, 0, 1);       // global_prefetch_b8
            v4f a0 = *(const v4f*)(ga);
            v4f a1 = *(const v4f*)(ga + 4);
            v8bf pk;
#pragma unroll
            for (int j = 0; j < 4; ++j) { pk[j] = (__bf16)a0[j]; pk[4 + j] = (__bf16)a1[j]; }
            *(v8bf*)(As + r * APAD + kc) = pk;
        }
#if HAVE_TDM
        const int cur = it & 1;
        if (wave == 0) {
            if (it + 1 < NT) {   // issue next tile, then wait for current (<=1 in flight)
                tdm_load_b_tile(B + (size_t)(k0 + BK) * N + nbase,
                                &Bf32[(cur ^ 1) * BK * BN], N, K);
                __builtin_amdgcn_s_wait_tensorcnt(1);
            } else {
                __builtin_amdgcn_s_wait_tensorcnt(0);
            }
        }
        __syncthreads();                              // current tile visible to all waves
        // convert Bf32[k][n] -> Bs[n][k] (bf16, transposed), packed dword stores
        {
            const int kk = (tid >> 4) * 2;            // even k
            const int n0 = (tid & 15) * 8;
            const float* r0 = &Bf32[cur * BK * BN + kk * BN + n0];
            const float* r1 = r0 + BN;
#pragma unroll
            for (int i = 0; i < 8; ++i)
                *(unsigned*)(&Bs[(n0 + i) * BPAD + kk]) = pack2bf(r0[i], r1[i]);
        }
#else
        // ---- stage B direct: pairs of k rows, packed dword LDS stores ----
        {
            const int kk = (tid >> 4) * 2;
            const int n0 = (tid & 15) * 8;
            const float* gb0 = B + (size_t)(k0 + kk) * N + nbase + n0;
            const float* gb1 = gb0 + N;
            __builtin_prefetch(gb0 + (size_t)BK * N, 0, 1);
            v4f b0a = *(const v4f*)(gb0), b0b = *(const v4f*)(gb0 + 4);
            v4f b1a = *(const v4f*)(gb1), b1b = *(const v4f*)(gb1 + 4);
#pragma unroll
            for (int i = 0; i < 4; ++i) {
                *(unsigned*)(&Bs[(n0 + i) * BPAD + kk])     = pack2bf(b0a[i], b1a[i]);
                *(unsigned*)(&Bs[(n0 + 4 + i) * BPAD + kk]) = pack2bf(b0b[i], b1b[i]);
            }
        }
#endif
        __syncthreads();

        v16bf af[2], bf[2];
#pragma unroll
        for (int mi = 0; mi < 2; ++mi) af[mi] = load_a_frag(As, wm * 32 + mi * 16, APAD);
#pragma unroll
        for (int ni = 0; ni < 2; ++ni) bf[ni] = load_b_frag(Bs, wn * 32 + ni * 16, BPAD);
#pragma unroll
        for (int mi = 0; mi < 2; ++mi)
#pragma unroll
            for (int ni = 0; ni < 2; ++ni)
                acc[mi][ni] = wmma_bf16(af[mi], bf[ni], acc[mi][ni]);
        __syncthreads();
    }

    // C layout: VGPR v -> row v (+8 for upper half-lanes), lane -> col.
    float* cb = C + (size_t)(mbase + wm * 32 + 8 * half) * N + (nbase + wn * 32 + nlan);
#pragma unroll
    for (int mi = 0; mi < 2; ++mi)
#pragma unroll
        for (int v = 0; v < 8; ++v)
#pragma unroll
            for (int ni = 0; ni < 2; ++ni)
                cb[(size_t)(mi * 16 + v) * N + ni * 16] = acc[mi][ni][v];
}

// ===================== rmsnorm (row-wise, in place) =====================
__global__ __launch_bounds__(256) void rmsnorm_kernel(float* __restrict__ x,
                                                      const float* __restrict__ w,
                                                      int L) {
    __shared__ float red[256];
    const int row = blockIdx.x, tid = threadIdx.x;
    v4f* xr = (v4f*)(x + (size_t)row * L);
    const v4f* wr = (const v4f*)w;
    const int L4 = L >> 2;
    float s = 0.f;
    for (int i = tid; i < L4; i += 256) {
        v4f v = xr[i];
        s += v[0] * v[0] + v[1] * v[1] + v[2] * v[2] + v[3] * v[3];
    }
    red[tid] = s;
    __syncthreads();
    for (int st = 128; st > 0; st >>= 1) {
        if (tid < st) red[tid] += red[tid + st];
        __syncthreads();
    }
    const float inv = rsqrtf(red[0] / (float)L + EPSF);
    for (int i = tid; i < L4; i += 256) {
        v4f v = xr[i], g = wr[i];
#pragma unroll
        for (int j = 0; j < 4; ++j) v[j] = v[j] * inv * g[j];
        xr[i] = v;
    }
}

// ===================== flash attention: 1 wave per (16-query tile, head) ===========
__global__ __launch_bounds__(32) void attn_kernel(const float* __restrict__ qkv,
                                                  float* __restrict__ out) {
    __shared__ __bf16 Kls[32 * KPAD];    // [key][d]   (B of Q@K^T)
    __shared__ __bf16 Vls[HDIM * VPAD];  // [d][key]   (B of P@V, transposed)
    __shared__ __bf16 Pls[16 * PPAD];    // P tile, pair-interleaved (p0->2c, p1->2c+1)
    const int lane = threadIdx.x;
    const int half = lane >> 4, nlan = lane & 15;
    const int qbase = blockIdx.x * 16;
    const int head  = blockIdx.y;
    const float scale = 0.08838834764831845f;  // 1/sqrt(128)

    const float* kptr = qkv + 1 * HID + head * HDIM;
    const float* vptr = qkv + 2 * HID + head * HDIM;

    // Q tile 16x128 -> 4 bf16 A-frags, vectorized f32x4 global reads
    v16bf qf[4];
    {
        const float* qp = qkv + (size_t)(qbase + nlan) * QKVW + head * HDIM;
#pragma unroll
        for (int kc = 0; kc < 4; ++kc) {
            const int kb = kc * 32 + half * 8;
            v4f q0 = *(const v4f*)(qp + kb);
            v4f q1 = *(const v4f*)(qp + kb + 4);
            v4f q2 = *(const v4f*)(qp + kb + 16);
            v4f q3 = *(const v4f*)(qp + kb + 20);
            v16bf f;
#pragma unroll
            for (int j = 0; j < 4; ++j) {
                f[j] = (__bf16)q0[j];  f[4 + j]  = (__bf16)q1[j];
                f[8 + j] = (__bf16)q2[j]; f[12 + j] = (__bf16)q3[j];
            }
            qf[kc] = f;
        }
    }

    v8f o[8] = {};
    float mrow[8], lrow[8];
#pragma unroll
    for (int v = 0; v < 8; ++v) { mrow[v] = -3.0e38f; lrow[v] = 0.f; }

    const int nkb = ((qbase + 15) >> 5) + 1;   // causal key blocks of 32
    for (int kb = 0; kb < nkb; ++kb) {
        const int keybase = kb * 32;
        // stage K (row-major, packed b64 stores) and V (transposed) tiles
        for (int i = 0; i < 32; ++i) {
            const int idx = i * 32 + lane;
            const int d4 = idx & 31, key = idx >> 5;
            v4f kv = *(const v4f*)(kptr + (size_t)(keybase + key) * QKVW + d4 * 4);
            v4f vv = *(const v4f*)(vptr + (size_t)(keybase + key) * QKVW + d4 * 4);
            v4bf kp;
#pragma unroll
            for (int j = 0; j < 4; ++j) kp[j] = (__bf16)kv[j];
            *(v4bf*)(&Kls[key * KPAD + d4 * 4]) = kp;
#pragma unroll
            for (int j = 0; j < 4; ++j) Vls[(d4 * 4 + j) * VPAD + key] = (__bf16)vv[j];
        }
        __syncthreads();

        // S = Q @ K^T : two 16x16 key subtiles
        v8f s0 = {}, s1 = {};
#pragma unroll
        for (int kc = 0; kc < 4; ++kc) {
            v16bf b0 = load_b_frag(Kls + kc * 32, 0, KPAD);
            v16bf b1 = load_b_frag(Kls + kc * 32, 16, KPAD);
            s0 = wmma_bf16(qf[kc], b0, s0);
            s1 = wmma_bf16(qf[kc], b1, s1);
        }

        // online softmax per row (row v+8*half lives in VGPR v, cols in lanes)
#pragma unroll
        for (int v = 0; v < 8; ++v) {
            const int grow = qbase + v + 8 * half;
            float e0 = s0[v] * scale, e1 = s1[v] * scale;
            if (keybase + nlan > grow) e0 = NEGINF;
            if (keybase + 16 + nlan > grow) e1 = NEGINF;
            const float mx = rmax16(fmaxf(e0, e1));
            const float mnew = fmaxf(mrow[v], mx);
            const float p0 = __expf(e0 - mnew), p1 = __expf(e1 - mnew);
            const float rs = rsum16(p0 + p1);
            const float alpha = __expf(mrow[v] - mnew);
            lrow[v] = lrow[v] * alpha + rs;
            mrow[v] = mnew;
#pragma unroll
            for (int n = 0; n < 8; ++n) o[n][v] *= alpha;
            // row r = v+8*half; logical col c -> slot 2c (first subtile), 2c+1 (second)
            *(unsigned*)(&Pls[(v + 8 * half) * PPAD + 2 * nlan]) = pack2bf(p0, p1);
        }
        __syncthreads();

        // O += P @ V : rebuild A-frag of P, de-interleaving the pair layout
        v16bf pa;
        {
            const __bf16* p = Pls + (size_t)nlan * PPAD;
            const int kbq = half * 8;
            v16bf f;
#pragma unroll
            for (int j = 0; j < 8; ++j) {
                f[j]     = p[2 * (kbq + j)];          // logical cols 0..15
                f[8 + j] = p[2 * (kbq + j) + 1];      // logical cols 16..31
            }
            pa = f;
        }
#pragma unroll
        for (int n = 0; n < 8; ++n) {
            v16bf vf = load_b_frag(Vls, n * 16, VPAD);
            o[n] = wmma_bf16(pa, vf, o[n]);
        }
        __syncthreads();
    }

    // normalize + store [SEQ, HID]; constant strides
    float* ob = out + (size_t)(qbase + 8 * half) * HID + head * HDIM + nlan;
    float rinv[8];
#pragma unroll
    for (int v = 0; v < 8; ++v) rinv[v] = 1.0f / lrow[v];
#pragma unroll
    for (int n = 0; n < 8; ++n)
#pragma unroll
        for (int v = 0; v < 8; ++v)
            ob[(size_t)v * HID + n * 16] = o[n][v] * rinv[v];
}

// ===================== MoE gate: softmax + top-2 (renormalized) =====================
__global__ __launch_bounds__(256) void gate_kernel(const float* __restrict__ h,
                                                   const float* __restrict__ wg,
                                                   int* __restrict__ top_i,
                                                   float* __restrict__ top_w) {
    __shared__ float red[256 * NEXP];
    const int t = blockIdx.x, tid = threadIdx.x;
    float loc[NEXP] = {};
    const float* hr = h + (size_t)t * HID;
    for (int k = tid; k < HID; k += 256) {
        const float x = hr[k];
        v4f g0 = *(const v4f*)(wg + k * NEXP);
        v4f g1 = *(const v4f*)(wg + k * NEXP + 4);
#pragma unroll
        for (int e = 0; e < 4; ++e) { loc[e] += x * g0[e]; loc[4 + e] += x * g1[e]; }
    }
#pragma unroll
    for (int e = 0; e < NEXP; ++e) red[tid * NEXP + e] = loc[e];
    __syncthreads();
    for (int st = 128; st > 0; st >>= 1) {
        if (tid < st)
#pragma unroll
            for (int e = 0; e < NEXP; ++e) red[tid * NEXP + e] += red[(tid + st) * NEXP + e];
        __syncthreads();
    }
    if (tid == 0) {
        float p[NEXP], mx = -3.0e38f, s = 0.f;
#pragma unroll
        for (int e = 0; e < NEXP; ++e) mx = fmaxf(mx, red[e]);
#pragma unroll
        for (int e = 0; e < NEXP; ++e) { p[e] = __expf(red[e] - mx); s += p[e]; }
#pragma unroll
        for (int e = 0; e < NEXP; ++e) p[e] /= s;
        int i0 = 0;
#pragma unroll
        for (int e = 1; e < NEXP; ++e) if (p[e] > p[i0]) i0 = e;
        int i1 = (i0 == 0) ? 1 : 0;
#pragma unroll
        for (int e = 0; e < NEXP; ++e) if (e != i0 && p[e] > p[i1]) i1 = e;
        const float ws = p[i0] + p[i1];
        top_i[t * 2 + 0] = i0;  top_w[t * 2 + 0] = p[i0] / ws;
        top_i[t * 2 + 1] = i1;  top_w[t * 2 + 1] = p[i1] / ws;
    }
}

// ===================== routing =====================
__global__ void zero_counts_kernel(int* __restrict__ counts) {
    if (threadIdx.x < NEXP) counts[threadIdx.x] = 0;
}
__global__ void route_kernel(const int* __restrict__ top_i, int* __restrict__ counts,
                             int* __restrict__ list) {
    const int slot = blockIdx.x * blockDim.x + threadIdx.x;  // 0..2047
    if (slot >= NSLOT) return;
    const int e = top_i[slot];
    const int pos = atomicAdd(&counts[e], 1);
    list[e * NSLOT + pos] = slot;   // slot = token*2 + k
}

// ===================== grouped expert GEMM (gathered rows, scaled per-slot store) ======
__global__ __launch_bounds__(256) void moe_gemm_kernel(const float* __restrict__ h,
                                                       const float* __restrict__ wexp,
                                                       const int* __restrict__ counts,
                                                       const int* __restrict__ list,
                                                       const float* __restrict__ tw,
                                                       float* __restrict__ eo) {
    __shared__ __bf16 As[BM * APAD];
    __shared__ __bf16 Bs[BN * BPAD];
#if HAVE_TDM
    __shared__ float Bf32[2 * BK * BN];  // double-buffered TDM landing tiles
#endif
    const int e = blockIdx.z;
    const int count = counts[e];
    const int mbase = blockIdx.y * BM;
    if (mbase >= count) return;
    const int nbase = blockIdx.x * BN;
    const int* lst = list + e * NSLOT;
    const float* B = wexp + (size_t)e * HID * HID;

    const int tid  = threadIdx.x;
    const int lane = tid & 31;
    const int wave = tid >> 5;
    const int wm = wave >> 2, wn = wave & 3;
    const int half = lane >> 4, nlan = lane & 15;
    constexpr int NT = HID / BK;

    v8f acc[2][2] = {};

#if HAVE_TDM
    if (wave == 0) tdm_load_b_tile(B + nbase, &Bf32[0], HID, HID);
#endif
    for (int it = 0; it < NT; ++it) {
        const int k0 = it * BK;
        {   // gather-stage A from routed tokens, packed b128 LDS store
            const int r = tid >> 2;
            const int kc = (tid & 3) * 8;
            const int grow = mbase + r;
            v8bf pk = {};
            if (grow < count) {
                const int token = lst[grow] >> 1;
                const float* ga = h + (size_t)token * HID + k0 + kc;
                v4f a0 = *(const v4f*)(ga);
                v4f a1 = *(const v4f*)(ga + 4);
#pragma unroll
                for (int j = 0; j < 4; ++j) { pk[j] = (__bf16)a0[j]; pk[4 + j] = (__bf16)a1[j]; }
            }
            *(v8bf*)(As + r * APAD + kc) = pk;
        }
#if HAVE_TDM
        const int cur = it & 1;
        if (wave == 0) {
            if (it + 1 < NT) {
                tdm_load_b_tile(B + (size_t)(k0 + BK) * HID + nbase,
                                &Bf32[(cur ^ 1) * BK * BN], HID, HID);
                __builtin_amdgcn_s_wait_tensorcnt(1);
            } else {
                __builtin_amdgcn_s_wait_tensorcnt(0);
            }
        }
        __syncthreads();
        {   // convert Bf32[k][n] -> Bs[n][k] (bf16, transposed)
            const int kk = (tid >> 4) * 2;
            const int n0 = (tid & 15) * 8;
            const float* r0 = &Bf32[cur * BK * BN + kk * BN + n0];
            const float* r1 = r0 + BN;
#pragma unroll
            for (int i = 0; i < 8; ++i)
                *(unsigned*)(&Bs[(n0 + i) * BPAD + kk]) = pack2bf(r0[i], r1[i]);
        }
#else
        {   // stage expert weight panel direct, transposed, packed dword stores
            const int kk = (tid >> 4) * 2;
            const int n0 = (tid & 15) * 8;
            const float* gb0 = B + (size_t)(k0 + kk) * HID + nbase + n0;
            const float* gb1 = gb0 + HID;
            __builtin_prefetch(gb0 + (size_t)BK * HID, 0, 1);
            v4f b0a = *(const v4f*)(gb0), b0b = *(const v4f*)(gb0 + 4);
            v4f b1a = *(const v4f*)(gb1), b1b = *(const v4f*)(gb1 + 4);
#pragma unroll
            for (int i = 0; i < 4; ++i) {
                *(unsigned*)(&Bs[(n0 + i) * BPAD + kk])     = pack2bf(b0a[i], b1a[i]);
                *(unsigned*)(&Bs[(n0 + 4 + i) * BPAD + kk]) = pack2bf(b0b[i], b1b[i]);
            }
        }
#endif
        __syncthreads();

        v16bf af[2], bf[2];
#pragma unroll
        for (int mi = 0; mi < 2; ++mi) af[mi] = load_a_frag(As, wm * 32 + mi * 16, APAD);
#pragma unroll
        for (int ni = 0; ni < 2; ++ni) bf[ni] = load_b_frag(Bs, wn * 32 + ni * 16, BPAD);
#pragma unroll
        for (int mi = 0; mi < 2; ++mi)
#pragma unroll
            for (int ni = 0; ni < 2; ++ni)
                acc[mi][ni] = wmma_bf16(af[mi], bf[ni], acc[mi][ni]);
        __syncthreads();
    }

    // per-slot scaled store (each slot owned by exactly one expert -> deterministic)
#pragma unroll
    for (int mi = 0; mi < 2; ++mi)
#pragma unroll
        for (int v = 0; v < 8; ++v) {
            const int grow = mbase + wm * 32 + mi * 16 + v + 8 * half;
            if (grow < count) {
                const int slot = lst[grow];
                const float w = tw[slot];
                float* eb = eo + (size_t)slot * HID + nbase + wn * 32 + nlan;
#pragma unroll
                for (int ni = 0; ni < 2; ++ni) eb[ni * 16] = w * acc[mi][ni][v];
            }
        }
}

// ===================== combine the two routed slots per token =====================
__global__ void combine_kernel(const float* __restrict__ eo, float* __restrict__ out) {
    const size_t i = (size_t)blockIdx.x * blockDim.x + threadIdx.x;  // v4f index
    if (i >= (size_t)SEQ * HID / 4) return;
    const size_t t = i / (HID / 4);
    const size_t c = i % (HID / 4);
    const v4f* a = (const v4f*)(eo + (t * 2 + 0) * HID);
    const v4f* b = (const v4f*)(eo + (t * 2 + 1) * HID);
    ((v4f*)(out + t * HID))[c] = a[c] + b[c];
}

// ===================== launch =====================
extern "C" void kernel_launch(void* const* d_in, const int* in_sizes, int n_in,
                              void* d_out, int out_size, void* d_ws, size_t ws_size,
                              hipStream_t stream) {
    const int*   input_ids = (const int*)d_in[0];
    const float* emb_table = (const float*)d_in[1];
    const float* w_qkv     = (const float*)d_in[2];
    const float* w_o       = (const float*)d_in[3];
    const float* norm1_w   = (const float*)d_in[4];
    const float* norm2_w   = (const float*)d_in[5];
    const float* w_gate    = (const float*)d_in[6];
    const float* w_experts = (const float*)d_in[7];
    float* out = (float*)d_out;

    char* ws = (char*)d_ws;
    const size_t MB = 1024ull * 1024ull;
    float* X    = (float*)(ws);                 // [SEQ,HID] embeddings, reused as h
    float* QKV  = (float*)(ws + 16 * MB);       // [SEQ,3*HID], reused as expert_out
    float* ATT  = (float*)(ws + 64 * MB);       // [SEQ,HID] attention output
    char*  SM   = ws + 80 * MB;                 // small buffers
    int*   top_i  = (int*)(SM);
    float* top_w  = (float*)(SM + NSLOT * sizeof(int));
    int*   counts = (int*)(SM + 2 * NSLOT * sizeof(int));
    int*   list   = (int*)(SM + 2 * NSLOT * sizeof(int) + 64);
    float* H  = X;      // h reuses embedding buffer (dead after QKV GEMM)
    float* EO = QKV;    // expert_out reuses qkv buffer (dead after attention)

    embed_kernel<<<SEQ, 256, 0, stream>>>(input_ids, emb_table, X);
    gemm_bf16_kernel<HID, QKVW><<<dim3(QKVW / BN, SEQ / BM), 256, 0, stream>>>(X, w_qkv, QKV);
    rmsnorm_kernel<<<SEQ, 256, 0, stream>>>(QKV, norm1_w, QKVW);
    attn_kernel<<<dim3(SEQ / 16, NHEADS), 32, 0, stream>>>(QKV, ATT);
    gemm_bf16_kernel<HID, HID><<<dim3(HID / BN, SEQ / BM), 256, 0, stream>>>(ATT, w_o, H);
    rmsnorm_kernel<<<SEQ, 256, 0, stream>>>(H, norm2_w, HID);
    gate_kernel<<<SEQ, 256, 0, stream>>>(H, w_gate, top_i, top_w);
    zero_counts_kernel<<<1, 32, 0, stream>>>(counts);
    route_kernel<<<NSLOT / 256, 256, 0, stream>>>(top_i, counts, list);
    moe_gemm_kernel<<<dim3(HID / BN, NSLOT / BM, NEXP), 256, 0, stream>>>(
        H, w_experts, counts, list, top_w, EO);
    combine_kernel<<<(SEQ * HID / 4 + 255) / 256, 256, 0, stream>>>(EO, out);
}
